// DeepseekV3TopkRouter_17325898072268
// MI455X (gfx1250) — compile-verified, matrix operand order checked by reading, not run on hardware
//
#include <hip/hip_runtime.h>
#include <hip/hip_bf16.h>

// ---------------- problem constants (match reference) ----------------
#define HID   7168
#define NEXP  256
#define TOPK  8
#define NGRP  8
#define TOPKG 4
#define SCALE 2.5f

// ---------------- GEMM tiling ----------------
// Block = 128 threads (4 waves). Each wave computes a 32x64 tile (2x4 wmma tiles),
// so per k-iteration: 8 v_wmma : 12 ds_load_b128 : 1 barrier.
#define TM    64     // tokens per block
#define TN    128    // experts per block (grid.y = 2)
#define KB    32     // k-step (one wmma k)
#define LDSW  40     // padded bf16 row stride (80B = 20 banks -> conflict-free)
#define NTHR  128

typedef __attribute__((ext_vector_type(16))) __bf16 v16bf;
typedef __attribute__((ext_vector_type(8)))  __bf16 v8bf;
typedef __attribute__((ext_vector_type(4)))  __bf16 v4bf;
typedef __attribute__((ext_vector_type(8)))  float  v8f;
typedef __attribute__((ext_vector_type(4)))  float  v4f;

static __device__ inline v4bf cvt4(v4f v) {
  v4bf o;
  o[0] = (__bf16)v[0]; o[1] = (__bf16)v[1];
  o[2] = (__bf16)v[2]; o[3] = (__bf16)v[3];
  return o;
}

struct Stage {   // per-thread register staging for one K-slab
  v4bf a[4];     // 64 rows x 8 chunks  / 128 thr = 4 chunks
  v4bf w[8];     // 128 rows x 8 chunks / 128 thr = 8 chunks
};

// ==================== Kernel 1: scores = sigmoid(hidden @ W^T + b) ====================
__global__ __launch_bounds__(NTHR)
void router_gemm_scores(const float* __restrict__ hidden,
                        const float* __restrict__ Wm,
                        const float* __restrict__ bvec,
                        float* __restrict__ scores)
{
  __shared__ __bf16 sA[2][TM * LDSW];   // 10 KB
  __shared__ __bf16 sB[2][TN * LDSW];   // 20 KB

  const int tid   = threadIdx.x;
  const int lane  = tid & 31;
  const int wave  = tid >> 5;
  const int wm    = wave >> 1;          // 0..1 : 32-row half
  const int wn    = wave & 1;           // 0..1 : 64-col half
  const int row0  = blockIdx.x * TM;
  const int col0  = blockIdx.y * TN;
  const int l16   = lane & 15;
  const int khalf = (lane & 16) >> 1;   // lane<16 -> K base 0 ; lane>=16 -> K base 8

  v8f acc[2][4];
  #pragma unroll
  for (int mi = 0; mi < 2; ++mi)
    #pragma unroll
    for (int ni = 0; ni < 4; ++ni)
      #pragma unroll
      for (int r = 0; r < 8; ++r) acc[mi][ni][r] = 0.0f;

  auto gload = [&](int k0) {
    Stage st;
    #pragma unroll
    for (int i = 0; i < 4; ++i) {       // A: 16B per chunk, coalesced per row
      int c = tid + i * NTHR;
      int r = c >> 3, kc = c & 7;
      v4f v = *(const v4f*)(hidden + (size_t)(row0 + r) * HID + k0 + kc * 4);
      st.a[i] = cvt4(v);
    }
    #pragma unroll
    for (int i = 0; i < 8; ++i) {       // W tile (L2-resident)
      int c = tid + i * NTHR;
      int r = c >> 3, kc = c & 7;
      v4f v = *(const v4f*)(Wm + (size_t)(col0 + r) * HID + k0 + kc * 4);
      st.w[i] = cvt4(v);
    }
    return st;
  };
  auto lstore = [&](int buf, const Stage& st) {
    #pragma unroll
    for (int i = 0; i < 4; ++i) {
      int c = tid + i * NTHR;
      int r = c >> 3, kc = c & 7;
      *(v4bf*)&sA[buf][r * LDSW + kc * 4] = st.a[i];
    }
    #pragma unroll
    for (int i = 0; i < 8; ++i) {
      int c = tid + i * NTHR;
      int r = c >> 3, kc = c & 7;
      *(v4bf*)&sB[buf][r * LDSW + kc * 4] = st.w[i];
    }
  };
  // Build a v16bf fragment matching the ISA 16-bit A/B layout:
  // per lane: 8 contiguous bf16 at K=khalf, 8 more at K=khalf+16.
  auto frag = [&](const __bf16* base, int row) {
    const __bf16* p = base + row * LDSW + khalf;
    v8bf lo = *(const v8bf*)(p);        // ds_load_b128
    v8bf hi = *(const v8bf*)(p + 16);   // ds_load_b128
    return __builtin_shufflevector(lo, hi, 0,1,2,3,4,5,6,7,8,9,10,11,12,13,14,15);
  };

  Stage st = gload(0);
  lstore(0, st);
  __syncthreads();

  const int NK = HID / KB;              // 224
  for (int it = 0; it < NK; ++it) {
    const int cur = it & 1;
    if (it + 1 < NK) st = gload((it + 1) * KB);   // prefetch next slab into regs

    v16bf af[2], bf[4];
    #pragma unroll
    for (int mi = 0; mi < 2; ++mi) af[mi] = frag(sA[cur], wm * 32 + mi * 16 + l16);
    #pragma unroll
    for (int ni = 0; ni < 4; ++ni) bf[ni] = frag(sB[cur], wn * 64 + ni * 16 + l16);

    #pragma unroll
    for (int mi = 0; mi < 2; ++mi)
      #pragma unroll
      for (int ni = 0; ni < 4; ++ni)
        acc[mi][ni] = __builtin_amdgcn_wmma_f32_16x16x32_bf16(
            false, af[mi], false, bf[ni], (short)0, acc[mi][ni], false, false);

    if (it + 1 < NK) lstore((it + 1) & 1, st);
    __syncthreads();
  }

  // Epilogue: + b, sigmoid, store fp32 scores (stays in L2: 8 MB total)
  const int lh = lane >> 4;
  #pragma unroll
  for (int mi = 0; mi < 2; ++mi) {
    #pragma unroll
    for (int ni = 0; ni < 4; ++ni) {
      const int col = col0 + wn * 64 + ni * 16 + l16;
      const float bb = bvec[col];
      #pragma unroll
      for (int r = 0; r < 8; ++r) {
        const int row = row0 + wm * 32 + mi * 16 + r + 8 * lh;  // C-matrix VGPR layout
        const float x = acc[mi][ni][r] + bb;
        scores[(size_t)row * NEXP + col] = 1.0f / (1.0f + __expf(-x));
      }
    }
  }
}

// ==================== Kernel 2: grouped top-k routing, one wave per token ====================
__global__ __launch_bounds__(256)
void router_topk(const float* __restrict__ scores,
                 const float* __restrict__ cbias,
                 int* __restrict__ idx_out,
                 float* __restrict__ w_out)
{
  const int lane = threadIdx.x & 31;
  const int t    = blockIdx.x * 8 + (threadIdx.x >> 5);
  const float NEG = -__builtin_inff();

  // lane owns 8 consecutive experts [8*lane, 8*lane+8) -> all in group lane>>2
  const float* sp = scores + (size_t)t * NEXP + lane * 8;
  v4f s0 = *(const v4f*)(sp);
  v4f s1 = *(const v4f*)(sp + 4);
  v4f c0 = *(const v4f*)(cbias + lane * 8);
  v4f c1 = *(const v4f*)(cbias + lane * 8 + 4);

  float s[8], sc[8];
  #pragma unroll
  for (int j = 0; j < 4; ++j) { s[j] = s0[j]; s[4 + j] = s1[j]; }
  #pragma unroll
  for (int j = 0; j < 4; ++j) { sc[j] = s[j] + c0[j]; sc[4 + j] = s[4 + j] + c1[j]; }

  // ---- per-group score = sum of top-2 of (score + bias) within the group of 32 experts ----
  float m1 = NEG, m2 = NEG;
  #pragma unroll
  for (int j = 0; j < 8; ++j) {
    float v = sc[j];
    if (v > m1) { m2 = m1; m1 = v; } else m2 = fmaxf(m2, v);
  }
  #pragma unroll
  for (int d = 1; d <= 2; d <<= 1) {    // merge top-2 pairs across the 4 lanes of a group
    float o1 = __shfl_xor(m1, d, 32);
    float o2 = __shfl_xor(m2, d, 32);
    float n1 = fmaxf(m1, o1);
    m2 = fmaxf(fminf(m1, o1), fmaxf(m2, o2));
    m1 = n1;
  }
  const float gscore = m1 + m2;

  float gs[NGRP];
  #pragma unroll
  for (int g = 0; g < NGRP; ++g) gs[g] = __shfl(gscore, g * 4, 32);

  const int g0 = lane >> 2;
  int rank = 0;
  #pragma unroll
  for (int h = 0; h < NGRP; ++h)
    rank += (gs[h] > gs[g0]) || ((gs[h] == gs[g0]) && (h < g0));   // top_k tie: lower index first
  const bool keep = rank < TOPKG;

  float mval[8];
  #pragma unroll
  for (int j = 0; j < 8; ++j) mval[j] = keep ? sc[j] : 0.0f;       // masked scores_for_choice

  // ---- iterative top-8 over 256 masked entries (carry raw sigmoid score for gating) ----
  int   top_idx[TOPK];
  float top_s[TOPK];
  #pragma unroll
  for (int k = 0; k < TOPK; ++k) {
    float bv = mval[0]; int bj = 0; float bs = s[0];
    #pragma unroll
    for (int j = 1; j < 8; ++j)
      if (mval[j] > bv) { bv = mval[j]; bj = j; bs = s[j]; }       // ties -> lower j
    int bidx = lane * 8 + bj;
    #pragma unroll
    for (int d = 16; d >= 1; d >>= 1) {                            // butterfly arg-max
      float ov   = __shfl_xor(bv, d, 32);
      int   oidx = __shfl_xor(bidx, d, 32);
      float os   = __shfl_xor(bs, d, 32);
      if ((ov > bv) || ((ov == bv) && (oidx < bidx))) { bv = ov; bidx = oidx; bs = os; }
    }
    top_idx[k] = bidx;
    top_s[k]   = bs;
    #pragma unroll
    for (int j = 0; j < 8; ++j)
      if (lane * 8 + j == bidx) mval[j] = NEG;                     // remove winner
  }

  float sum = 1e-20f;
  #pragma unroll
  for (int k = 0; k < TOPK; ++k) sum += top_s[k];
  const float inv = SCALE / sum;

  if (lane == 0) {
    #pragma unroll
    for (int k = 0; k < TOPK; ++k) {
      idx_out[(size_t)t * TOPK + k] = top_idx[k];
      w_out[(size_t)t * TOPK + k]   = top_s[k] * inv;
    }
  }
}

// ==================== launcher ====================
extern "C" void kernel_launch(void* const* d_in, const int* in_sizes, int n_in,
                              void* d_out, int out_size, void* d_ws, size_t ws_size,
                              hipStream_t stream) {
  const float* hidden = (const float*)d_in[0];   // [T, 7168] fp32
  const float* Wm     = (const float*)d_in[1];   // [256, 7168] fp32
  const float* bvec   = (const float*)d_in[2];   // [256] fp32
  const float* cbias  = (const float*)d_in[3];   // [256] fp32

  const int T = in_sizes[0] / HID;               // 8192
  float* scores = (float*)d_ws;                  // T*256 fp32 = 8 MB, L2-resident

  // Output tuple flattened: [T*8] int32 indices, then [T*8] fp32 weights.
  int*   idx_out = (int*)d_out;
  float* w_out   = (float*)d_out + (size_t)T * TOPK;

  dim3 g1(T / TM, NEXP / TN);
  router_gemm_scores<<<g1, NTHR, 0, stream>>>(hidden, Wm, bvec, scores);
  router_topk<<<T / 8, 256, 0, stream>>>(scores, cbias, idx_out, w_out);
}
